// MultiAttentionBlock_43456479101353
// MI455X (gfx1250) — matrix operand order repaired and checked by reading, not compiled, not run on hardware
//
#include <hip/hip_runtime.h>
#include <hip/hip_bf16.h>
#include <math.h>

typedef __attribute__((ext_vector_type(16))) _Float16 v16h;
typedef __attribute__((ext_vector_type(8)))  _Float16 v8h;
typedef __attribute__((ext_vector_type(2)))  _Float16 v2h;
typedef __attribute__((ext_vector_type(8)))  float    v8f;

#define HEADS 12
#define CH 64
#define HC 768   // HEADS*CH
#define NEG_SLOPE 0.2f
#define GAT_EPS 1e-16f

// ---------- helpers: order-preserving float <-> uint for atomic max ----------
__device__ __forceinline__ unsigned f2ord(float f) {
    int i = __float_as_int(f);
    return (i >= 0) ? ((unsigned)i + 0x80000000u) : ~(unsigned)i;
}
__device__ __forceinline__ float ord2f(unsigned u) {
    int i = (u & 0x80000000u) ? (int)(u - 0x80000000u) : (int)~u;
    return __int_as_float(i);
}

// ---------- phase 0: conversions / packing / init ----------
__global__ void k_convert_x(const float* __restrict__ x, _Float16* __restrict__ xh, int n) {
    int i = blockIdx.x * blockDim.x + threadIdx.x;
    if (i < n) xh[i] = (_Float16)x[i];
}

// Pack W[64,768] (row-major, K-major rows) into the WMMA B-operand order used here (contiguous k, same as A):
// per (ntile, ktile, lane): 16 contiguous f16 = K run for that lane's column.
// lane<16 -> column n = nt*16+lane, K = kt*32 + 0..15
// lane>=16 -> column n = nt*16+lane-16, K = kt*32 + 16..31
__global__ void k_pack_w(const float* __restrict__ W, _Float16* __restrict__ bp) {
    int idx = blockIdx.x * blockDim.x + threadIdx.x;   // [0, 48*2*32*16)
    if (idx >= 48 * 2 * 32 * 16) return;
    int i    = idx & 15;
    int lane = (idx >> 4) & 31;
    int kt   = (idx >> 9) & 1;
    int nt   = idx >> 10;
    int k = kt * 32 + ((lane >> 4) << 4) + i;
    int n = nt * 16 + (lane & 15);
    bp[idx] = (_Float16)W[k * HC + n];
}

__global__ void k_init(float* __restrict__ out, unsigned* __restrict__ menc,
                       float* __restrict__ denom, int N) {
    int i = blockIdx.x * blockDim.x + threadIdx.x;
    if (i < N * CH) out[i] = 0.0f;
    if (i < N * HEADS) { menc[i] = 0u; denom[i] = 0.0f; }
}

// ---------- phase 1: h = x @ W via v_wmma_f32_16x16x32_f16 ----------
// One wave computes a 16x64 strip of h (4 N-tiles, K=64 in 2 WMMA steps each).
__global__ void k_gemm(const _Float16* __restrict__ xh, const _Float16* __restrict__ bp,
                       _Float16* __restrict__ h, int N, int Mtiles) {
    int gwave = (int)((blockIdx.x * (size_t)blockDim.x + threadIdx.x) >> 5);
    int lane  = threadIdx.x & 31;
    int mt = gwave / 12;          // 12 groups of 64 output columns
    int ng = gwave % 12;
    if (mt >= Mtiles) return;

    int m0  = mt * 16;
    int row = m0 + (lane & 15);
    if (row >= N) row = N - 1;    // clamp loads; stores guarded below
    int n0 = ng * 64;

    v8f cacc[4] = {};
    #pragma unroll
    for (int kt = 0; kt < 2; ++kt) {
        // A operand layout (contiguous, matching packed B):
        //   lane<16 : K = kt*32 + {0..15}
        //   lane>=16: K = kt*32 + {16..31}
        int kbase = kt * 32 + ((lane < 16) ? 0 : 16);
        const v8h* ap = (const v8h*)(xh + (size_t)row * CH + kbase);
        v8h alo = ap[0];
        v8h ahi = ap[1];          // +8 halves
        v16h a;
        #pragma unroll
        for (int i = 0; i < 8; ++i) { a[i] = alo[i]; a[i + 8] = ahi[i]; }

        #pragma unroll
        for (int j = 0; j < 4; ++j) {
            int nt = (n0 >> 4) + j;
            v16h b = *(const v16h*)(bp + ((((size_t)nt * 2 + kt) * 32 + lane) << 4));
            cacc[j] = __builtin_amdgcn_wmma_f32_16x16x32_f16(
                false, a, false, b, (short)0, cacc[j], false, false);
        }
    }

    // D layout: VGPR v -> row m0 + v + (lane<16?0:8), col = n0 + j*16 + lane%16
    int mbase = m0 + ((lane < 16) ? 0 : 8);
    int ncol  = lane & 15;
    #pragma unroll
    for (int v = 0; v < 8; ++v) {
        int m = mbase + v;
        if (m < N) {
            _Float16* hp = h + (size_t)m * HC + n0 + ncol;
            hp[0]  = (_Float16)cacc[0][v];
            hp[16] = (_Float16)cacc[1][v];
            hp[32] = (_Float16)cacc[2][v];
            hp[48] = (_Float16)cacc[3][v];
        }
    }
}

// ---------- phase 2: per-(node,head) attention scores ----------
__global__ void k_attn(const _Float16* __restrict__ h, const float* __restrict__ att_src,
                       const float* __restrict__ att_dst, float* __restrict__ a_s,
                       float* __restrict__ a_d, int N) {
    int idx = blockIdx.x * blockDim.x + threadIdx.x;
    if (idx >= N * HEADS) return;
    int hd = idx % HEADS, n = idx / HEADS;
    const _Float16* hp = h + (size_t)n * HC + hd * CH;
    const float* as = att_src + hd * CH;
    const float* ad = att_dst + hd * CH;
    float s = 0.f, d = 0.f;
    #pragma unroll 8
    for (int c = 0; c < CH; ++c) {
        float v = (float)hp[c];
        s += v * as[c];
        d += v * ad[c];
    }
    a_s[idx] = s;
    a_d[idx] = d;
}

// ---------- phase 3: segment max (softmax numerator stabilization) ----------
__global__ void k_edge_max(const int* __restrict__ ei, const float* __restrict__ a_s,
                           const float* __restrict__ a_d, unsigned* __restrict__ menc,
                           int E, int N) {
    int idx = blockIdx.x * blockDim.x + threadIdx.x;
    if (idx >= (E + N) * HEADS) return;
    int hd = idx % HEADS, e = idx / HEADS;
    int s, d;
    if (e < E) { s = ei[e]; d = ei[E + e]; } else { s = d = e - E; }
    float t = a_s[s * HEADS + hd] + a_d[d * HEADS + hd];
    t = (t > 0.f) ? t : NEG_SLOPE * t;
    atomicMax(&menc[d * HEADS + hd], f2ord(t));
}

// ---------- phase 4: segment sum of exp ----------
__global__ void k_edge_denom(const int* __restrict__ ei, const float* __restrict__ a_s,
                             const float* __restrict__ a_d, const unsigned* __restrict__ menc,
                             float* __restrict__ denom, int E, int N) {
    int idx = blockIdx.x * blockDim.x + threadIdx.x;
    if (idx >= (E + N) * HEADS) return;
    int hd = idx % HEADS, e = idx / HEADS;
    int s, d;
    if (e < E) { s = ei[e]; d = ei[E + e]; } else { s = d = e - E; }
    float t = a_s[s * HEADS + hd] + a_d[d * HEADS + hd];
    t = (t > 0.f) ? t : NEG_SLOPE * t;
    float ex = __expf(t - ord2f(menc[d * HEADS + hd]));
    atomicAdd(&denom[d * HEADS + hd], ex);
}

// ---------- phase 5: fused weighted aggregation + head mean ----------
// One wave per edge; lane l owns output columns 2l, 2l+1; head-mean folded in.
__global__ void k_edge_agg(const int* __restrict__ ei, const float* __restrict__ a_s,
                           const float* __restrict__ a_d, const unsigned* __restrict__ menc,
                           const float* __restrict__ denom, const _Float16* __restrict__ h,
                           float* __restrict__ out, int E, int N) {
    int gwave = (int)((blockIdx.x * (size_t)blockDim.x + threadIdx.x) >> 5);
    int lane  = threadIdx.x & 31;
    if (gwave >= E + N) return;
    int e = gwave, s, d;
    if (e < E) { s = ei[e]; d = ei[E + e]; } else { s = d = e - E; }

    float w = 0.f;
    if (lane < HEADS) {
        int a = d * HEADS + lane;
        float t = a_s[s * HEADS + lane] + a_d[a];
        t = (t > 0.f) ? t : NEG_SLOPE * t;
        w = __expf(t - ord2f(menc[a])) / (denom[a] + GAT_EPS);
    }

    float acc0 = 0.f, acc1 = 0.f;
    const _Float16* hp = h + (size_t)s * HC + lane * 2;
    #pragma unroll
    for (int hd = 0; hd < HEADS; ++hd) {
        float wh = __shfl(w, hd, 32);
        v2h hv = *(const v2h*)(hp + hd * CH);
        acc0 += wh * (float)hv[0];
        acc1 += wh * (float)hv[1];
    }
    const float inv = 1.0f / (float)HEADS;
    float* op = out + (size_t)d * CH + lane * 2;
    atomicAdd(op,     acc0 * inv);
    atomicAdd(op + 1, acc1 * inv);
}

// ---------- phase 6: residual + bias + ReLU (in place on d_out) ----------
__global__ void k_finalize(const float* __restrict__ x, const float* __restrict__ bias,
                           float* __restrict__ out, int N) {
    int i = blockIdx.x * blockDim.x + threadIdx.x;
    if (i >= N * CH) return;
    float v = x[i] + out[i] + bias[i % CH];
    out[i] = v > 0.f ? v : 0.f;
}

extern "C" void kernel_launch(void* const* d_in, const int* in_sizes, int n_in,
                              void* d_out, int out_size, void* d_ws, size_t ws_size,
                              hipStream_t stream) {
    const float* x       = (const float*)d_in[0];
    const int*   ei      = (const int*)d_in[1];
    const float* W       = (const float*)d_in[2];
    const float* att_src = (const float*)d_in[3];
    const float* att_dst = (const float*)d_in[4];
    const float* bias    = (const float*)d_in[5];
    int N = in_sizes[0] / CH;
    int E = in_sizes[1] / 2;

    // workspace carve-out (256B aligned)
    char* ws = (char*)d_ws;
    size_t off = 0;
    auto carve = [&](size_t bytes) -> char* {
        char* p = ws + off;
        off = (off + bytes + 255) & ~(size_t)255;
        return p;
    };
    _Float16* h     = (_Float16*)carve((size_t)N * HC * sizeof(_Float16));
    _Float16* xh    = (_Float16*)carve((size_t)N * CH * sizeof(_Float16));
    _Float16* bpk   = (_Float16*)carve((size_t)48 * 2 * 32 * 16 * sizeof(_Float16));
    float*    a_s   = (float*)carve((size_t)N * HEADS * sizeof(float));
    float*    a_d   = (float*)carve((size_t)N * HEADS * sizeof(float));
    unsigned* menc  = (unsigned*)carve((size_t)N * HEADS * sizeof(unsigned));
    float*    denom = (float*)carve((size_t)N * HEADS * sizeof(float));
    float*    out   = (float*)d_out;

    const int thr = 256;
    auto blks = [&](long n) { return (int)((n + thr - 1) / thr); };

    k_convert_x<<<blks((long)N * CH), thr, 0, stream>>>(x, xh, N * CH);
    k_pack_w<<<blks(48 * 2 * 32 * 16), thr, 0, stream>>>(W, bpk);
    k_init<<<blks((long)N * CH), thr, 0, stream>>>(out, menc, denom, N);

    int Mtiles = (N + 15) / 16;
    long gemm_waves = (long)Mtiles * 12;
    k_gemm<<<blks(gemm_waves * 32), thr, 0, stream>>>(xh, bpk, h, N, Mtiles);

    k_attn<<<blks((long)N * HEADS), thr, 0, stream>>>(h, att_src, att_dst, a_s, a_d, N);

    long etot = (long)(E + N) * HEADS;
    k_edge_max<<<blks(etot), thr, 0, stream>>>(ei, a_s, a_d, menc, E, N);
    k_edge_denom<<<blks(etot), thr, 0, stream>>>(ei, a_s, a_d, menc, denom, E, N);

    long agg_waves = (long)(E + N);
    k_edge_agg<<<blks(agg_waves * 32), thr, 0, stream>>>(ei, a_s, a_d, menc, denom, h, out, E, N);

    k_finalize<<<blks((long)N * CH), thr, 0, stream>>>(x, bias, out, N);
}
